// HeteroRGCN_6133213298982
// MI455X (gfx1250) — matrix operand order repaired and checked by reading, not compiled, not run on hardware
//
#include <hip/hip_runtime.h>

typedef __attribute__((ext_vector_type(16))) _Float16 v16h;
typedef __attribute__((ext_vector_type(8)))  _Float16 v8h;
typedef __attribute__((ext_vector_type(8)))  float    v8f;

#define DIN  128
#define DH   128
#define DOUT 64
#define RNUM 4

union V16U { v16h v; v8h h[2]; };

// ---------------- utility kernels ----------------

__global__ void zero_f32_kernel(float* __restrict__ p, long n) {
  long t = (long)blockIdx.x * blockDim.x + threadIdx.x;
  if (t < n) p[t] = 0.0f;
}

__global__ void cvt_f16_kernel(const float* __restrict__ s, _Float16* __restrict__ d, long n) {
  long t = (long)blockIdx.x * blockDim.x + threadIdx.x;
  if (t < n) d[t] = (_Float16)s[t];
}

__global__ void relu_cvt_kernel(const float* __restrict__ s, _Float16* __restrict__ d, long n) {
  long t = (long)blockIdx.x * blockDim.x + threadIdx.x;
  if (t < n) d[t] = (_Float16)fmaxf(s[t], 0.0f);
}

__global__ void degree_kernel(const int* __restrict__ dst, float* __restrict__ deg, int E) {
  int t = blockIdx.x * blockDim.x + threadIdx.x;
  if (t < E) atomicAdd(deg + dst[t], 1.0f);
}

__global__ void recip_kernel(float* __restrict__ deg, long n) {
  long t = (long)blockIdx.x * blockDim.x + threadIdx.x;
  if (t < n) deg[t] = 1.0f / fmaxf(deg[t], 1.0f);
}

// Pack W[r][k][n] (f32, row-major K x NOUT) into per-lane WMMA B-fragment order:
// layout [r][kt][nt][lane(32)][half(16)], element (k,n) with
//   n = nt*16 + (lane & 15), k = kt*32 + (lane>>4)*8 + (half&7) + (half>>3)*16
template<int NOUT>
__global__ void pack_w_kernel(const float* __restrict__ W, _Float16* __restrict__ Wp) {
  constexpr int KT = DIN / 32;
  constexpr int NT = NOUT / 16;
  const long total = (long)RNUM * KT * NT * 512;
  long t = (long)blockIdx.x * blockDim.x + threadIdx.x;
  if (t >= total) return;
  const int  h    = (int)(t & 15);
  const int  lane = (int)((t >> 4) & 31);
  const long frag = t >> 9;
  const int  nt   = (int)(frag % NT);
  const long f2   = frag / NT;
  const int  kt   = (int)(f2 % KT);
  const int  r    = (int)(f2 / KT);
  const int  n    = nt * 16 + (lane & 15);
  const int  k    = kt * 32 + (lane >> 4) * 8 + (h & 7) + (h >> 3) * 16;
  Wp[t] = (_Float16)W[((long)r * DIN + k) * NOUT + n];
}

// ---------------- WMMA GEMM: C[N,NOUT] = A[N,128] * Wpacked + bias ----------------
// One wave32 computes a 16-row strip across all NOUT columns.
template<int NOUT>
__global__ void __launch_bounds__(256)
gemm_wmma_kernel(const _Float16* __restrict__ A,   // [N,128] f16 row-major
                 const _Float16* __restrict__ Wp,  // packed [4][NOUT/16][512]
                 const float*    __restrict__ bias,// [NOUT]
                 float*          __restrict__ C,   // [N,NOUT] f32
                 int N) {
  constexpr int NT = NOUT / 16;
  const int lane   = threadIdx.x & 31;
  const int wave   = threadIdx.x >> 5;
  const int row0   = (blockIdx.x * 8 + wave) * 16;
  if (row0 + 16 > N) return;           // uniform per-wave exit: EXEC all-ones at WMMA
  const int laneHi = lane >> 4;
  const int m      = lane & 15;

  // A fragments: 16-bit A 16x32 layout; lane holds K = {kb..kb+7, kb+16..kb+23},
  // kb = 32*kt + 8*laneHi  -> two contiguous 16B loads per fragment.
  const _Float16* arow = A + (size_t)(row0 + m) * DIN;
  v16h a[4];
#pragma unroll
  for (int kt = 0; kt < 4; ++kt) {
    const int kb = kt * 32 + laneHi * 8;
    V16U u;
    u.h[0] = *(const v8h*)(arow + kb);
    u.h[1] = *(const v8h*)(arow + kb + 16);
    a[kt] = u.v;
  }

  const int col_in = lane & 15;
#pragma unroll
  for (int nt = 0; nt < NT; ++nt) {
    v8f acc = {};
#pragma unroll
    for (int kt = 0; kt < 4; ++kt) {
      const _Float16* bp = Wp + (((kt * NT + nt) << 9) + lane * 16);
      V16U u;
      u.h[0] = *(const v8h*)bp;
      u.h[1] = *(const v8h*)(bp + 8);
      acc = __builtin_amdgcn_wmma_f32_16x16x32_f16(
          /*neg_a=*/false, a[kt], /*neg_b=*/false, u.v,
          /*c_mod=*/(short)0, acc, /*reuse_a=*/false, /*reuse_b=*/false);
    }
    const int col = nt * 16 + col_in;
    const float bs = bias[col];
    // C layout: VGPR g -> row = row0 + laneHi*8 + g, col = lane&15 (+16*nt)
    float* cbase = C + (size_t)(row0 + laneHi * 8) * NOUT + col;
#pragma unroll
    for (int g = 0; g < 8; ++g) cbase[(size_t)g * NOUT] = acc[g] + bs;
  }
}

// ---------------- edge scatter with fused mean ----------------
template<int D>
__global__ void scatter_mean_kernel(const float* __restrict__ Wh,
                                    const int*   __restrict__ src,
                                    const int*   __restrict__ dst,
                                    const float* __restrict__ invdeg,
                                    float*       __restrict__ out,
                                    int E) {
  constexpr int LG = (D == 128) ? 7 : 6;
  long t = (long)blockIdx.x * blockDim.x + threadIdx.x;
  if (t >= ((long)E << LG)) return;
  const int c = (int)(t & (D - 1));
  const int e = (int)(t >> LG);
  const int s = src[e];
  const int d = dst[e];
  const float v = Wh[((size_t)s << LG) + c] * invdeg[d];
  atomicAdd(out + ((size_t)d << LG) + c, v);
}

// ---------------- launcher ----------------
extern "C" void kernel_launch(void* const* d_in, const int* in_sizes, int n_in,
                              void* d_out, int out_size, void* d_ws, size_t ws_size,
                              hipStream_t stream) {
  (void)n_in; (void)out_size; (void)ws_size;
  const float* feat  = (const float*)d_in[0];
  const float* W1    = (const float*)d_in[1];
  const float* b1    = (const float*)d_in[2];
  const float* W2    = (const float*)d_in[3];
  const float* b2    = (const float*)d_in[4];
  const int*   edges = (const int*)d_in[5];
  float*       out   = (float*)d_out;

  const int N = in_sizes[0] / DIN;
  const int E = in_sizes[5] / (RNUM * 2);

  // workspace carve-out
  char* ws = (char*)d_ws;
  size_t off = 0;
  auto alloc = [&](size_t bytes) -> char* {
    char* p = ws + off;
    off += (bytes + 255) & ~(size_t)255;
    return p;
  };
  _Float16* feat_h = (_Float16*)alloc((size_t)N * DIN * sizeof(_Float16));
  _Float16* hr_h   = (_Float16*)alloc((size_t)N * DH  * sizeof(_Float16));
  float*    hacc   = (float*)   alloc((size_t)N * DH  * sizeof(float));
  float*    Wh     = (float*)   alloc((size_t)N * DH  * sizeof(float));   // reused per etype & layer
  float*    invdeg = (float*)   alloc((size_t)RNUM * N * sizeof(float));
  _Float16* Wp1    = (_Float16*)alloc((size_t)RNUM * 4 * (DH   / 16) * 512 * sizeof(_Float16));
  _Float16* Wp2    = (_Float16*)alloc((size_t)RNUM * 4 * (DOUT / 16) * 512 * sizeof(_Float16));

  const int B = 256;
  auto g1 = [&](long n) -> unsigned { return (unsigned)((n + B - 1) / B); };

  // init accumulators / outputs
  zero_f32_kernel<<<g1((long)N * DH),   B, 0, stream>>>(hacc,   (long)N * DH);
  zero_f32_kernel<<<g1((long)RNUM * N), B, 0, stream>>>(invdeg, (long)RNUM * N);
  zero_f32_kernel<<<g1((long)N * DOUT), B, 0, stream>>>(out,    (long)N * DOUT);

  // precision conversion + weight packing
  cvt_f16_kernel<<<g1((long)N * DIN), B, 0, stream>>>(feat, feat_h, (long)N * DIN);
  pack_w_kernel<DH>  <<<g1((long)RNUM * 4 * (DH   / 16) * 512), B, 0, stream>>>(W1, Wp1);
  pack_w_kernel<DOUT><<<g1((long)RNUM * 4 * (DOUT / 16) * 512), B, 0, stream>>>(W2, Wp2);

  // degrees (shared by both layers: same edge lists)
  for (int r = 0; r < RNUM; ++r) {
    const int* dst = edges + ((size_t)r * 2 + 1) * E;
    degree_kernel<<<g1(E), B, 0, stream>>>(dst, invdeg + (size_t)r * N, E);
  }
  recip_kernel<<<g1((long)RNUM * N), B, 0, stream>>>(invdeg, (long)RNUM * N);

  const unsigned gemmGrid = (unsigned)((N / 16 + 7) / 8);

  // layer 1: per-etype GEMM -> gather/mean-scatter, summed into hacc
  for (int r = 0; r < RNUM; ++r) {
    const int* src = edges + (size_t)r * 2 * E;
    const int* dst = src + E;
    gemm_wmma_kernel<DH><<<gemmGrid, 256, 0, stream>>>(
        feat_h, Wp1 + (size_t)r * 4 * (DH / 16) * 512, b1 + (size_t)r * DH, Wh, N);
    scatter_mean_kernel<DH><<<g1((long)E * DH), B, 0, stream>>>(
        Wh, src, dst, invdeg + (size_t)r * N, hacc, E);
  }

  // relu + downconvert for layer-2 A operand
  relu_cvt_kernel<<<g1((long)N * DH), B, 0, stream>>>(hacc, hr_h, (long)N * DH);

  // layer 2: per-etype GEMM -> scatter directly into d_out
  for (int r = 0; r < RNUM; ++r) {
    const int* src = edges + (size_t)r * 2 * E;
    const int* dst = src + E;
    gemm_wmma_kernel<DOUT><<<gemmGrid, 256, 0, stream>>>(
        hr_h, Wp2 + (size_t)r * 4 * (DOUT / 16) * 512, b2 + (size_t)r * DOUT, Wh, N);
    scatter_mean_kernel<DOUT><<<g1((long)E * DOUT), B, 0, stream>>>(
        Wh, src, dst, invdeg + (size_t)r * N, out, E);
  }
}
